// LocalFeatureTransformer_15015205667475
// MI455X (gfx1250) — matrix operand order repaired and checked by reading, not compiled, not run on hardware
//
#include <hip/hip_runtime.h>
#include <hip/hip_bf16.h>

typedef __bf16 bf16_t;
typedef __attribute__((ext_vector_type(16))) __bf16 v16bf;
typedef __attribute__((ext_vector_type(8)))  float  v8f;

__device__ __forceinline__ int imin(int a, int b) { return a < b ? a : b; }
__device__ __forceinline__ int imax(int a, int b) { return a > b ? a : b; }

// ---- WMMA (CDNA5 wave32, 16x16x32 bf16 -> f32) ----
__device__ __forceinline__ v8f wmma_bf16(v16bf a, v16bf b, v8f c) {
  return __builtin_amdgcn_wmma_f32_16x16x32_bf16(false, a, false, b, (short)0, c,
                                                 false, false);
}

// Fragment layouts (per 16x16x32 WMMA tile), 512 bf16 per fragment:
//   frag[lane*16 + e]
// A (16x32 MxK):  m = lane&15, hi = lane>>4, k = (e&7) + 8*hi + 16*(e>>3)
//   inverse: given k: hiA = (k>>3)&1, e = (k&7) + ((k&16)>>1)
// B (32x16 KxN):  n = lane&15, hi = lane>>4, k = (e&7) + 8*hi + 16*(e>>3)
//   inverse: given k: hiB = (k>>3)&1, e = (k&7) + ((k&16)>>1)
__device__ __forceinline__ v16bf frag_ld_g(const bf16_t* p) {
  return *(const v16bf*)p;  // 32B per lane -> 2x global_load_b128, coalesced
}
__device__ __forceinline__ v16bf frag_ld_s(const bf16_t* p) {
  return *(const v16bf*)p;  // 32B per lane -> 2x ds_load_b128
}

// ---------------- layout / conversion kernels ----------------
__global__ void k_nchw2nhwc(const float* __restrict__ in, float* __restrict__ out) {
  int i = blockIdx.x * 256 + threadIdx.x;
  int c = i & 255;
  int r = i >> 8;
  int x = r % 96; r /= 96;
  int y = r % 96; int b = r / 96;
  out[i] = in[((b * 256 + c) * 96 + y) * 96 + x];
}

__global__ void k_nhwc2nchw(const float* __restrict__ in, float* __restrict__ out) {
  int i = blockIdx.x * 256 + threadIdx.x;
  int x = i % 96;
  int r = i / 96;
  int y = r % 96; r /= 96;
  int c = r % 256; int b = r / 256;
  out[i] = in[((b * 96 + y) * 96 + x) * 256 + c];
}

// fp32 [L][K][N] -> bf16 B-fragment-major [L][kk][nt][lane][16]
__global__ void k_cvt_swz(const float* __restrict__ src, bf16_t* __restrict__ dst,
                          int K, int N, int total) {
  int i = blockIdx.x * 256 + threadIdx.x;
  if (i >= total) return;
  int kn = K * N;
  int layer = i / kn, r = i % kn;
  int k = r / N, n = r % N;
  int kk = k >> 5, kr = k & 31;
  int hi = (kr >> 3) & 1, e = (kr & 7) + ((kr & 16) >> 1);
  int nt = n >> 4, nlo = n & 15;
  int l = hi * 16 + nlo;
  dst[(size_t)layer * kn + ((size_t)(kk * (N >> 4) + nt) * 32 + l) * 16 + e] =
      (bf16_t)src[i];
}

// RoPE tables [576][256]
__global__ void k_rope_tables(float* __restrict__ st, float* __restrict__ ct) {
  int pos = blockIdx.x;
  int p = threadIdx.x;
  int y = pos / 24, x = pos % 24;
  int u = p >> 1;
  int t = u >> 1;
  const float LOG1E4_D4 = 0.14391156831212787f;  // ln(10000)/64
  float dv = __expf(-(float)t * LOG1E4_D4);
  float coord = (u & 1) ? (float)(x + 1) : (float)(y + 1);
  st[pos * 256 + p] = __sinf(coord * dv);
  ct[pos * 256 + p] = __cosf(coord * dv);
}

// ---------------- aggregation + LayerNorm -> A-fragment layout ----------------
// mode 0: depthwise 4x4/4 conv, mode 1: 4x4 maxpool; then LN over C=256.
// out: bf16 A-frag layout [mtile(72)][kk(8)][lane][16]
__global__ void k_agg(const float* __restrict__ X, const float* __restrict__ aggw,
                      const float* __restrict__ g, const float* __restrict__ bt,
                      bf16_t* __restrict__ out, int mode) {
  __shared__ float wred[8];
  __shared__ float smu, svar;
  int t = blockIdx.x;
  int c = threadIdx.x;
  int b = t / 576, pos = t % 576, oy = pos / 24, ox = pos % 24;
  float val;
  if (mode == 0) {
    val = 0.f;
#pragma unroll
    for (int dy = 0; dy < 4; ++dy)
#pragma unroll
      for (int dx = 0; dx < 4; ++dx)
        val += X[((b * 96 + oy * 4 + dy) * 96 + ox * 4 + dx) * 256 + c] *
               aggw[c * 16 + dy * 4 + dx];
  } else {
    val = -3.0e38f;
#pragma unroll
    for (int dy = 0; dy < 4; ++dy)
#pragma unroll
      for (int dx = 0; dx < 4; ++dx)
        val = fmaxf(val, X[((b * 96 + oy * 4 + dy) * 96 + ox * 4 + dx) * 256 + c]);
  }
  float s = val;
  for (int o = 16; o > 0; o >>= 1) s += __shfl_xor(s, o, 32);
  if ((threadIdx.x & 31) == 0) wred[threadIdx.x >> 5] = s;
  __syncthreads();
  if (threadIdx.x == 0) {
    float tt = 0.f;
    for (int i = 0; i < 8; ++i) tt += wred[i];
    smu = tt * (1.f / 256.f);
  }
  __syncthreads();
  float d = val - smu;
  float q = d * d;
  for (int o = 16; o > 0; o >>= 1) q += __shfl_xor(q, o, 32);
  if ((threadIdx.x & 31) == 0) wred[threadIdx.x >> 5] = q;
  __syncthreads();
  if (threadIdx.x == 0) {
    float tt = 0.f;
    for (int i = 0; i < 8; ++i) tt += wred[i];
    svar = tt * (1.f / 256.f);
  }
  __syncthreads();
  float inv = rsqrtf(svar + 1e-5f);
  float ov = d * inv * g[c] + bt[c];
  // A-frag store
  int mt = t >> 4, m = t & 15;
  int kk = c >> 5, kr = c & 31;
  int hiA = (kr >> 3) & 1, e = (kr & 7) + ((kr & 16) >> 1);
  int l = hiA * 16 + m;
  out[((size_t)(mt * 8 + kk) * 32 + l) * 16 + e] = (bf16_t)ov;
}

// ---------------- 1152x256x256 GEMM, zero LDS, fragment-direct ----------------
// 2x M-blocking: each block owns 32 rows, so every B fragment feeds 2 WMMAs.
// mode: 0 = Q-frag out, 1 = K-frag(B^T) out, 2 = V-frag(B) out, 3 = f32 row-major
__global__ void k_gemm256(const bf16_t* __restrict__ A, const bf16_t* __restrict__ B,
                          void* __restrict__ Out, const float* __restrict__ rs,
                          const float* __restrict__ rc, int mode, int rope) {
  int lane = threadIdx.x & 31, w = threadIdx.x >> 5;
  int mt0 = blockIdx.x * 2;
  v8f acc[2][2] = {};
#pragma unroll
  for (int kk = 0; kk < 8; ++kk) {
    v16bf a0 = frag_ld_g(A + ((size_t)((mt0 + 0) * 8 + kk) * 32 + lane) * 16);
    v16bf a1 = frag_ld_g(A + ((size_t)((mt0 + 1) * 8 + kk) * 32 + lane) * 16);
#pragma unroll
    for (int j = 0; j < 2; ++j) {
      int nt = w * 2 + j;
      v16bf bb = frag_ld_g(B + ((size_t)(kk * 16 + nt) * 32 + lane) * 16);
      acc[0][j] = wmma_bf16(a0, bb, acc[0][j]);
      acc[1][j] = wmma_bf16(a1, bb, acc[1][j]);
    }
  }
  int hic = lane >> 4, nlo = lane & 15;
#pragma unroll
  for (int m2 = 0; m2 < 2; ++m2) {
#pragma unroll
    for (int j = 0; j < 2; ++j) {
      int col = (w * 2 + j) * 16 + nlo;
#pragma unroll
      for (int v = 0; v < 8; ++v) {
        int row = (mt0 + m2) * 16 + v + 8 * hic;
        float val = acc[m2][j][v];
        if (rope) {  // out[p] = x[p]*cos + sign*x[p^1]*sin (lane pairs hold col^1)
          int pos = row % 576;
          float partner = __shfl_xor(val, 1, 32);
          float sgn = (col & 1) ? 1.f : -1.f;
          val = val * rc[pos * 256 + col] + sgn * partner * rs[pos * 256 + col];
        }
        if (mode == 3) {
          ((float*)Out)[row * 256 + col] = val;
        } else {
          int b = row / 576, pos = row % 576;
          int h = col >> 5, cc = col & 31;
          size_t dst;
          if (mode == 0) {          // Q: A-frag per (b,h,ltile)
            int lt = pos >> 4, m = pos & 15;
            int hiA = (cc >> 3) & 1, e = (cc & 7) + ((cc & 16) >> 1);
            dst = ((size_t)((b * 8 + h) * 36 + lt) * 32 + hiA * 16 + m) * 16 + e;
          } else if (mode == 1) {   // K: B^T-frag per (b,h,stile)
            int st = pos >> 4, n = pos & 15;
            int hiB = (cc >> 3) & 1, e = (cc & 7) + ((cc & 16) >> 1);
            dst = ((size_t)((b * 8 + h) * 36 + st) * 32 + hiB * 16 + n) * 16 + e;
          } else {                  // V: B-frag per (b,h,kstep,ntile)
            int ks = pos >> 5, kr = pos & 31;
            int hiB = (kr >> 3) & 1, e = (kr & 7) + ((kr & 16) >> 1);
            int ntv = cc >> 4, n = cc & 15;
            dst = ((size_t)(((b * 8 + h) * 18 + ks) * 2 + ntv) * 32 + hiB * 16 + n) * 16 + e;
          }
          ((bf16_t*)Out)[dst] = (bf16_t)val;
        }
      }
    }
  }
}

// ---------------- attention per (Ltile, head, batch) ----------------
// Qf: A-frags [2][8][36][512]; Kf: B^T-frags [2][8][36][512];
// Vf: B-frags [2][8][18][2][512]; O: A-frag layout for the merge GEMM.
__global__ void k_attn(const bf16_t* __restrict__ Qf, const bf16_t* __restrict__ Kf,
                       const bf16_t* __restrict__ Vf, bf16_t* __restrict__ O) {
  __shared__ __attribute__((aligned(32))) float  sc[16 * 576];   // 36 KB scores
  __shared__ __attribute__((aligned(32))) bf16_t pr[18 * 512];   // 18 KB probs (A-frag)
  int lt = blockIdx.x, h = blockIdx.y, b = blockIdx.z;
  int lane = threadIdx.x & 31, w = threadIdx.x >> 5;
  int hic = lane >> 4, nlo = lane & 15;
  const float scale = 0.17677669529663687f;  // 1/sqrt(32)

  // scores: K dim = dh = 32 -> one WMMA per 16-wide S tile
  v16bf a = frag_ld_g(Qf + ((size_t)((b * 8 + h) * 36 + lt) * 32 + lane) * 16);
  for (int st = w; st < 36; st += 8) {
    v16bf bb = frag_ld_g(Kf + ((size_t)((b * 8 + h) * 36 + st) * 32 + lane) * 16);
    v8f c = {};
    c = wmma_bf16(a, bb, c);
#pragma unroll
    for (int v = 0; v < 8; ++v)
      sc[(v + 8 * hic) * 576 + st * 16 + nlo] = c[v] * scale;
  }
  __syncthreads();

  // softmax (2 rows per wave), probs stored as A-fragments
  for (int r = w * 2; r < w * 2 + 2; ++r) {
    float mx = -3.0e38f;
    for (int cc = lane; cc < 576; cc += 32) mx = fmaxf(mx, sc[r * 576 + cc]);
    for (int o = 16; o > 0; o >>= 1) mx = fmaxf(mx, __shfl_xor(mx, o, 32));
    float sm = 0.f;
    for (int cc = lane; cc < 576; cc += 32) {
      float e = __expf(sc[r * 576 + cc] - mx);
      sc[r * 576 + cc] = e;
      sm += e;
    }
    for (int o = 16; o > 0; o >>= 1) sm += __shfl_xor(sm, o, 32);
    float inv = 1.f / sm;
    for (int cc = lane; cc < 576; cc += 32) {
      int ks = cc >> 5, kr = cc & 31;
      int hiA = (kr >> 3) & 1, e = (kr & 7) + ((kr & 16) >> 1);
      pr[ks * 512 + (hiA * 16 + r) * 16 + e] = (bf16_t)(sc[r * 576 + cc] * inv);
    }
  }
  __syncthreads();

  // probs @ V : 18 K-steps; waves 0,1 each own one 16-col N tile
  if (w < 2) {
    v8f accv = {};
    for (int ks = 0; ks < 18; ++ks) {
      v16bf a2 = frag_ld_s(pr + ks * 512 + lane * 16);
      v16bf b2 = frag_ld_g(Vf + ((size_t)(((b * 8 + h) * 18 + ks) * 2 + w) * 32 + lane) * 16);
      accv = wmma_bf16(a2, b2, accv);
    }
    // write A-frag layout (input of merge GEMM)
#pragma unroll
    for (int v = 0; v < 8; ++v) {
      int row = b * 576 + lt * 16 + v + 8 * hic;
      int col = h * 32 + w * 16 + nlo;
      int mt = row >> 4, m = row & 15;
      int kk = col >> 5, kr = col & 31;
      int hiA = (kr >> 3) & 1, e = (kr & 7) + ((kr & 16) >> 1);
      O[((size_t)(mt * 8 + kk) * 32 + hiA * 16 + m) * 16 + e] = (bf16_t)accv[v];
    }
  }
}

// ---------------- fused MLP: cat->w1->leaky->w2->LN + residual ----------------
// 32 tokens per workgroup (2 row-tiles): every weight fragment feeds 2 WMMAs.
__global__ void k_mlp(float* __restrict__ Xio, const float* __restrict__ Msm,
                      const bf16_t* __restrict__ W1, const bf16_t* __restrict__ W2,
                      const float* __restrict__ g2, const float* __restrict__ b2) {
  __shared__ __attribute__((aligned(32))) bf16_t sA[32 * 512];  // A-frags [kk(16)][m2(2)][512], 32 KB
  __shared__ __attribute__((aligned(32))) bf16_t sH[32 * 512];  // hidden A-frags, 32 KB
  float* sC = (float*)sA;  // 32x256 f32 out tile, reuses sA after GEMM1

  int tid = threadIdx.x, lane = tid & 31, w = tid >> 5;
  int tok0 = blockIdx.x * 32;
  int b = tok0 / 9216, rem = tok0 % 9216, Y = rem / 96, X0 = rem % 96;

  float fy = (Y + 0.5f) * 0.25f - 0.5f;
  int y0 = (int)floorf(fy);
  float wy = fy - (float)y0;
  int y0c = imin(imax(y0, 0), 23), y1c = imin(imax(y0 + 1, 0), 23);

  // build A = [x | upsample4x(m)] as 32 A-fragments
  for (int idx = tid; idx < 32 * 512; idx += 256) {
    int r = idx >> 9, cc = idx & 511;   // r = token row 0..31
    int tok = tok0 + r;
    float v;
    if (cc < 256) {
      v = Xio[tok * 256 + cc];
    } else {
      int ch = cc - 256;
      int Xc = X0 + r;
      float fx = (Xc + 0.5f) * 0.25f - 0.5f;
      int x0 = (int)floorf(fx);
      float wx = fx - (float)x0;
      int x0c = imin(imax(x0, 0), 23), x1c = imin(imax(x0 + 1, 0), 23);
      const float* Mb = Msm + b * 576 * 256;
      float v00 = Mb[(y0c * 24 + x0c) * 256 + ch];
      float v01 = Mb[(y0c * 24 + x1c) * 256 + ch];
      float v10 = Mb[(y1c * 24 + x0c) * 256 + ch];
      float v11 = Mb[(y1c * 24 + x1c) * 256 + ch];
      v = (1.f - wy) * ((1.f - wx) * v00 + wx * v01) +
          wy * ((1.f - wx) * v10 + wx * v11);
    }
    int m2 = r >> 4, rr = r & 15;
    int kk = cc >> 5, kr = cc & 31;
    int hiA = (kr >> 3) & 1, e = (kr & 7) + ((kr & 16) >> 1);
    sA[(kk * 2 + m2) * 512 + (hiA * 16 + rr) * 16 + e] = (bf16_t)v;
  }
  __syncthreads();

  // GEMM1: (32x512) @ W1(512x512); wave owns 4 N-tiles x 2 M-tiles
  v8f acc[2][4] = {};
  for (int kk = 0; kk < 16; ++kk) {
    v16bf a0 = frag_ld_s(sA + (kk * 2 + 0) * 512 + lane * 16);
    v16bf a1 = frag_ld_s(sA + (kk * 2 + 1) * 512 + lane * 16);
#pragma unroll
    for (int t4 = 0; t4 < 4; ++t4) {
      int nt = w * 4 + t4;
      v16bf bb = frag_ld_g(W1 + ((size_t)(kk * 32 + nt) * 32 + lane) * 16);
      acc[0][t4] = wmma_bf16(a0, bb, acc[0][t4]);
      acc[1][t4] = wmma_bf16(a1, bb, acc[1][t4]);
    }
  }

  // LeakyReLU -> hidden A-fragments
  int hic = lane >> 4, nlo = lane & 15;
#pragma unroll
  for (int m2 = 0; m2 < 2; ++m2) {
#pragma unroll
    for (int t4 = 0; t4 < 4; ++t4) {
      int col = (w * 4 + t4) * 16 + nlo;
      int kk2 = col >> 5, kr = col & 31;
      int hiA = (kr >> 3) & 1, e = (kr & 7) + ((kr & 16) >> 1);
#pragma unroll
      for (int v = 0; v < 8; ++v) {
        float vf = acc[m2][t4][v];
        vf = vf >= 0.f ? vf : 0.01f * vf;
        sH[(kk2 * 2 + m2) * 512 + (hiA * 16 + v + 8 * hic) * 16 + e] = (bf16_t)vf;
      }
    }
  }
  __syncthreads();

  // GEMM2: (32x512) @ W2(512x256); wave owns 2 N-tiles x 2 M-tiles
  v8f a2[2][2] = {};
  for (int kk = 0; kk < 16; ++kk) {
    v16bf a0 = frag_ld_s(sH + (kk * 2 + 0) * 512 + lane * 16);
    v16bf a1 = frag_ld_s(sH + (kk * 2 + 1) * 512 + lane * 16);
#pragma unroll
    for (int j = 0; j < 2; ++j) {
      int nt = w * 2 + j;
      v16bf bb = frag_ld_g(W2 + ((size_t)(kk * 16 + nt) * 32 + lane) * 16);
      a2[0][j] = wmma_bf16(a0, bb, a2[0][j]);
      a2[1][j] = wmma_bf16(a1, bb, a2[1][j]);
    }
  }
  __syncthreads();  // all sA reads done before aliased sC writes

  // dump C tile (32x256), LayerNorm per row, residual add
#pragma unroll
  for (int m2 = 0; m2 < 2; ++m2) {
#pragma unroll
    for (int j = 0; j < 2; ++j) {
      int col = (w * 2 + j) * 16 + nlo;
#pragma unroll
      for (int v = 0; v < 8; ++v)
        sC[(m2 * 16 + v + 8 * hic) * 256 + col] = a2[m2][j][v];
    }
  }
  __syncthreads();
  for (int r = w * 4; r < w * 4 + 4; ++r) {
    float s = 0.f, sq = 0.f;
    for (int cc = lane; cc < 256; cc += 32) {
      float v = sC[r * 256 + cc];
      s += v; sq += v * v;
    }
    for (int o = 16; o > 0; o >>= 1) {
      s += __shfl_xor(s, o, 32);
      sq += __shfl_xor(sq, o, 32);
    }
    float mu = s * (1.f / 256.f);
    float var = sq * (1.f / 256.f) - mu * mu;
    float inv = rsqrtf(var + 1e-5f);
    int tok = tok0 + r;
    for (int cc = lane; cc < 256; cc += 32) {
      float v = (sC[r * 256 + cc] - mu) * inv * g2[cc] + b2[cc];
      Xio[tok * 256 + cc] += v;
    }
  }
}

// ---------------- host orchestration ----------------
extern "C" void kernel_launch(void* const* d_in, const int* in_sizes, int n_in,
                              void* d_out, int out_size, void* d_ws, size_t ws_size,
                              hipStream_t stream) {
  (void)in_sizes; (void)n_in; (void)out_size; (void)ws_size;
  const float* feat0 = (const float*)d_in[0];
  const float* feat1 = (const float*)d_in[1];
  const float* aggw  = (const float*)d_in[2];
  const float* qw    = (const float*)d_in[3];
  const float* kw    = (const float*)d_in[4];
  const float* vw    = (const float*)d_in[5];
  const float* mw    = (const float*)d_in[6];
  const float* w1    = (const float*)d_in[7];
  const float* w2    = (const float*)d_in[8];
  const float* ln1g  = (const float*)d_in[9];
  const float* ln1b  = (const float*)d_in[10];
  const float* ln2g  = (const float*)d_in[11];
  const float* ln2b  = (const float*)d_in[12];

  char* p = (char*)d_ws;
  auto carve = [&](size_t bytes) {
    void* r = (void*)p;
    p += (bytes + 255) & ~(size_t)255;
    return r;
  };
  float*  ropeS = (float*)carve(576 * 256 * 4);
  float*  ropeC = (float*)carve(576 * 256 * 4);
  float*  xb0   = (float*)carve((size_t)4718592 * 4);
  float*  xb1   = (float*)carve((size_t)4718592 * 4);
  float*  msm   = (float*)carve(1152 * 256 * 4);
  bf16_t* qln   = (bf16_t*)carve(1152 * 256 * 2);
  bf16_t* sln   = (bf16_t*)carve(1152 * 256 * 2);
  bf16_t* qp    = (bf16_t*)carve(1152 * 256 * 2);
  bf16_t* kp    = (bf16_t*)carve(1152 * 256 * 2);
  bf16_t* vp    = (bf16_t*)carve(1152 * 256 * 2);
  bf16_t* ao    = (bf16_t*)carve(1152 * 256 * 2);
  bf16_t* wqb   = (bf16_t*)carve((size_t)8 * 65536 * 2);
  bf16_t* wkb   = (bf16_t*)carve((size_t)8 * 65536 * 2);
  bf16_t* wvb   = (bf16_t*)carve((size_t)8 * 65536 * 2);
  bf16_t* wmb   = (bf16_t*)carve((size_t)8 * 65536 * 2);
  bf16_t* w1b   = (bf16_t*)carve((size_t)8 * 262144 * 2);
  bf16_t* w2b   = (bf16_t*)carve((size_t)8 * 131072 * 2);

  k_nchw2nhwc<<<18432, 256, 0, stream>>>(feat0, xb0);
  k_nchw2nhwc<<<18432, 256, 0, stream>>>(feat1, xb1);
  k_rope_tables<<<576, 256, 0, stream>>>(ropeS, ropeC);
  k_cvt_swz<<<2048, 256, 0, stream>>>(qw, wqb, 256, 256, 8 * 65536);
  k_cvt_swz<<<2048, 256, 0, stream>>>(kw, wkb, 256, 256, 8 * 65536);
  k_cvt_swz<<<2048, 256, 0, stream>>>(vw, wvb, 256, 256, 8 * 65536);
  k_cvt_swz<<<2048, 256, 0, stream>>>(mw, wmb, 256, 256, 8 * 65536);
  k_cvt_swz<<<8192, 256, 0, stream>>>(w1, w1b, 512, 512, 8 * 262144);
  k_cvt_swz<<<4096, 256, 0, stream>>>(w2, w2b, 512, 256, 8 * 131072);

  for (int i = 0; i < 8; ++i) {
    bool self = (i % 2) == 0;
    auto run = [&](float* x, float* src, int rope) {
      k_agg<<<1152, 256, 0, stream>>>(x,   aggw + i * 4096, ln1g + i * 256, ln1b + i * 256, qln, 0);
      k_agg<<<1152, 256, 0, stream>>>(src, aggw + i * 4096, ln1g + i * 256, ln1b + i * 256, sln, 1);
      k_gemm256<<<36, 256, 0, stream>>>(qln, wqb + i * 65536, (void*)qp, ropeS, ropeC, 0, rope);
      k_gemm256<<<36, 256, 0, stream>>>(sln, wkb + i * 65536, (void*)kp, ropeS, ropeC, 1, rope);
      k_gemm256<<<36, 256, 0, stream>>>(sln, wvb + i * 65536, (void*)vp, ropeS, ropeC, 2, 0);
      k_attn<<<dim3(36, 8, 2), 256, 0, stream>>>(qp, kp, vp, ao);
      k_gemm256<<<36, 256, 0, stream>>>(ao, wmb + i * 65536, (void*)msm, ropeS, ropeC, 3, 0);
      k_mlp<<<576, 256, 0, stream>>>(x, msm, w1b + i * 262144, w2b + i * 131072,
                                     ln2g + i * 256, ln2b + i * 256);
    };
    if (self) {
      run(xb0, xb0, 1);
      run(xb1, xb1, 1);
    } else {
      run(xb0, xb1, 0);   // cross: feat0 <- (feat0, feat1)
      run(xb1, xb0, 0);   // cross: feat1 <- (feat1, updated feat0)
    }
  }

  k_nhwc2nchw<<<18432, 256, 0, stream>>>(xb0, (float*)d_out);
  k_nhwc2nchw<<<18432, 256, 0, stream>>>(xb1, (float*)d_out + 4718592);
}
